// ScaledDotProductAttention_20169166422395
// MI455X (gfx1250) — compile-verified, matrix operand order checked by reading, not run on hardware
//
#include <hip/hip_runtime.h>
#include <hip/hip_bf16.h>

// ---------------------------------------------------------------------------
// Flash-attention for MI455X (gfx1250, wave32, WMMA + TDM async tile staging).
//   scores = softmax(mask ? -inf : (Q K^T)/sqrt(64)) ; out = scores @ V
//   BH=32, L=2048, D=64. Matmuls in bf16 WMMA (16x16x32), softmax/accum fp32.
//
// Fast path (needs 16 MB of d_ws): pre-pass converts K -> bf16 row-major and
// V -> bf16 transposed ([bh][dim][key]); main kernel stages K/V/mask tiles
// with tensor_load_to_lds (double-buffered, s_wait_tensorcnt) so all inner-
// loop global traffic is DMA'd while WMMAs run.  Fallback: VMEM staging.
// ---------------------------------------------------------------------------

typedef __attribute__((ext_vector_type(16))) __bf16       v16bf;
typedef __attribute__((ext_vector_type(8)))  float        v8f;
typedef __attribute__((ext_vector_type(4)))  unsigned int v4u;
typedef __attribute__((ext_vector_type(8)))  int          v8i32;
typedef __attribute__((ext_vector_type(4)))  int          v4i32;

#define BH_N    32
#define SEQ     2048
#define DH      64
#define QTILE   64      // query rows per block (4 waves x 16)
#define KTILE   64      // key rows per iteration
#define NTILES  (SEQ / KTILE)
#define NWAVE   4
#define THREADS (NWAVE * 32)
#define LSTRIDE 72      // bf16 LDS row stride: 64 elems + TDM pad (4 dwords)
#define MSTRIDE 68      // mask LDS row stride: 64 B + TDM pad (1 dword)

#define ATTN_SCALE 0.125f   // 1/sqrt(64)

__device__ __forceinline__ float red_max16(float v) {
#pragma unroll
    for (int off = 1; off < 16; off <<= 1)
        v = fmaxf(v, __shfl_xor(v, off, 32));
    return v;
}
__device__ __forceinline__ float red_sum16(float v) {
#pragma unroll
    for (int off = 1; off < 16; off <<= 1)
        v += __shfl_xor(v, off, 32);
    return v;
}

// ---------------------------------------------------------------------------
// TDM: issue a 2D tile load (global -> LDS) via descriptor groups (ISA 8.3-8.6)
//   group0: count=1 | lds_addr | global_addr(57b) | type=2
//   group1: data_size, pad_enable/interval/amount, tensor dims, tile dims,
//           tensor_dim0_stride (all in data_size units)
// clang-23 / therock form: 6 args (g0, g1, g2, g3, extra group, cpol)
// ---------------------------------------------------------------------------
__device__ __forceinline__ void tdm_load_2d(const void* gtile, unsigned lds_off,
                                            unsigned ds_code,   // 0=1B,1=2B,2=4B
                                            unsigned pad_int,   // interval code
                                            unsigned pad_amt,   // amount code
                                            unsigned tensor_d0, unsigned tensor_d1,
                                            unsigned tile_d0,   unsigned tile_d1,
                                            unsigned stride0)
{
    const unsigned long long ga = (unsigned long long)gtile;
    v4u g0;
    g0[0] = 1u;                                          // count=1, user desc
    g0[1] = lds_off;                                     // lds_addr (bytes)
    g0[2] = (unsigned)(ga & 0xffffffffu);                // global_addr[31:0]
    g0[3] = (unsigned)((ga >> 32) & 0x01ffffffu)         // global_addr[56:32]
          | (2u << 30);                                  // type = 2 ("image")
    v8i32 g1;
    g1[0] = (int)((ds_code << 16) | (1u << 20)           // pad_enable
                | (pad_int << 22) | (pad_amt << 25));
    g1[1] = (int)(tensor_d0 << 16);                      // tensor_dim0[15:0]
    g1[2] = (int)((tensor_d0 >> 16) | (tensor_d1 << 16));
    g1[3] = (int)((tensor_d1 >> 16) | (tile_d0 << 16));
    g1[4] = (int)(tile_d1 & 0xffffu);                    // tile_dim1, tile_dim2=0
    g1[5] = (int)stride0;                                // tensor_dim0_stride lo
    g1[6] = 0;
    g1[7] = 0;
    const v4i32 z4 = {0, 0, 0, 0};                       // groups 2/3 unused (2D)
    const v8i32 z8 = {0, 0, 0, 0, 0, 0, 0, 0};
    __builtin_amdgcn_tensor_load_to_lds(g0, g1, z4, z4, z8, 0);
}

// ---------------------------------------------------------------------------
// Pre-pass: K -> bf16 (row-major), V -> bf16 transposed [bh][dim][key]
// ---------------------------------------------------------------------------
__global__ __launch_bounds__(256)
void cvt_kv_kernel(const float* __restrict__ K, const float* __restrict__ V,
                   __bf16* __restrict__ Kb, __bf16* __restrict__ VbT)
{
    const size_t i = (size_t)blockIdx.x * blockDim.x + threadIdx.x;
    const size_t n = (size_t)BH_N * SEQ * DH;
    if (i >= n) return;
    Kb[i] = (__bf16)K[i];
    const int    d  = (int)(i & (DH - 1));
    const size_t l  = (i >> 6) & (SEQ - 1);
    const size_t bh = i >> 17;                            // / (SEQ*DH)
    VbT[bh * (size_t)(DH * SEQ) + (size_t)d * SEQ + l] = (__bf16)V[i];
}

// ---------------------------------------------------------------------------
// Main kernel: TDM-staged flash attention
// ---------------------------------------------------------------------------
__global__ __launch_bounds__(THREADS)
void fa_wmma_tdm_kernel(const float* __restrict__ Q,
                        const __bf16* __restrict__ Kb,
                        const __bf16* __restrict__ VbT,
                        const unsigned char* __restrict__ M,
                        float* __restrict__ O)
{
    __shared__ __bf16        Kt[2][KTILE * LSTRIDE];      // [key][dim]
    __shared__ __bf16        Vt[2][DH * LSTRIDE];         // [dim][key] (transposed)
    __shared__ unsigned char Mt[2][QTILE * MSTRIDE];      // [qrow][key]
    __shared__ __bf16        Pl[NWAVE][16 * LSTRIDE];     // per-wave P staging

    const int tid  = threadIdx.x;
    const int wave = tid >> 5;
    const int lane = tid & 31;
    const int m    = lane & 15;
    const int kh   = lane >> 4;

    const int qtile = blockIdx.x;
    const int bh    = blockIdx.y;
    const int qbase = qtile * QTILE + wave * 16;

    const float*         Qg = Q   + (size_t)bh * SEQ * DH;
    const __bf16*        Kg = Kb  + (size_t)bh * SEQ * DH;
    const __bf16*        Vg = VbT + (size_t)bh * DH * SEQ;
    const unsigned char* Mg = M   + (size_t)bh * SEQ * SEQ
                                  + (size_t)(qtile * QTILE) * SEQ;

    // ---- TDM tile issue (wave 0 only; EXEC-independent, wave-level op) -----
    auto issue_tile = [&](int kt, int buf) {
        const int kb = kt * KTILE;
        // K tile: 64 keys x 64 dims, bf16, row 32 dwords + 4 dword pad -> 72
        tdm_load_2d(Kg + (size_t)kb * DH,
                    (unsigned)(unsigned long long)&Kt[buf][0],
                    1u, 4u, 3u, DH, SEQ, DH, KTILE, DH);
        // V^T tile: 64 dims x 64 keys, bf16, same padding
        tdm_load_2d(Vg + kb,
                    (unsigned)(unsigned long long)&Vt[buf][0],
                    1u, 4u, 3u, SEQ, DH, KTILE, DH, SEQ);
        // mask tile: 64 qrows x 64 keys, 1B, row 16 dwords + 1 dword pad -> 68
        tdm_load_2d(Mg + kb,
                    (unsigned)(unsigned long long)&Mt[buf][0],
                    0u, 3u, 0u, SEQ, SEQ, KTILE, QTILE, SEQ);
    };

    if (wave == 0) issue_tile(0, 0);

    // ---- Q fragments (A layout, 16x32 each; K=0..31 and K=32..63) ----------
    v16bf qa0, qa1;
    {
        const float* qr = Qg + (size_t)(qbase + m) * DH;
#pragma unroll
        for (int i = 0; i < 8; ++i) {
            qa0[i]     = (__bf16)qr[      kh * 8 + i];
            qa0[8 + i] = (__bf16)qr[16 +  kh * 8 + i];
            qa1[i]     = (__bf16)qr[32 +  kh * 8 + i];
            qa1[8 + i] = (__bf16)qr[48 +  kh * 8 + i];
        }
    }

    float mrun[8], lrun[8];
    v8f   o[4];
    const v8f vzero = (v8f){0.f, 0.f, 0.f, 0.f, 0.f, 0.f, 0.f, 0.f};
#pragma unroll
    for (int r = 0; r < 8; ++r) { mrun[r] = -__builtin_inff(); lrun[r] = 0.f; }
#pragma unroll
    for (int t = 0; t < 4; ++t) o[t] = vzero;

    __bf16* pst = &Pl[wave][0];

    for (int kt = 0; kt < NTILES; ++kt) {
        const int cur = kt & 1;

        // double-buffer: issue next tile, wait for current (TDM is in-order,
        // so tensorcnt<=3 after issuing 3 => the previous 3 have completed)
        if (wave == 0) {
            if (kt + 1 < NTILES) {
                issue_tile(kt + 1, cur ^ 1);
                __builtin_amdgcn_s_wait_tensorcnt(3);
            } else {
                __builtin_amdgcn_s_wait_tensorcnt(0);
            }
        }
        __syncthreads();                       // tile kt visible to all waves

        const __bf16*        Kl = &Kt[cur][0];
        const __bf16*        Vl = &Vt[cur][0];
        const unsigned char* Ml = &Mt[cur][0];

        // ---- S = Q K^T : four 16x16 score tiles ----------------------------
        v8f sc[4];
#pragma unroll
        for (int j = 0; j < 4; ++j) {
            v16bf kb0, kb1;
            const __bf16* kp = Kl + (j * 16 + m) * LSTRIDE;   // n = lane&15
#pragma unroll
            for (int i = 0; i < 8; ++i) {
                kb0[i]     = kp[      kh * 8 + i];
                kb0[8 + i] = kp[16 +  kh * 8 + i];
                kb1[i]     = kp[32 +  kh * 8 + i];
                kb1[8 + i] = kp[48 +  kh * 8 + i];
            }
            v8f acc = vzero;
            acc = __builtin_amdgcn_wmma_f32_16x16x32_bf16(false, qa0, false, kb0,
                                                          (short)0, acc, false, false);
            acc = __builtin_amdgcn_wmma_f32_16x16x32_bf16(false, qa1, false, kb1,
                                                          (short)0, acc, false, false);
            sc[j] = acc;
        }

        // ---- scale + mask (from LDS mask tile) -----------------------------
        const unsigned char* mrow = Ml + (wave * 16 + kh * 8) * MSTRIDE;
#pragma unroll
        for (int j = 0; j < 4; ++j) {
#pragma unroll
            for (int r = 0; r < 8; ++r) {
                const unsigned char mk = mrow[r * MSTRIDE + j * 16 + m];
                const float s = sc[j][r] * ATTN_SCALE;
                sc[j][r] = mk ? -__builtin_inff() : s;
            }
        }

        // ---- online softmax ------------------------------------------------
        float mnew[8], alpha[8];
#pragma unroll
        for (int r = 0; r < 8; ++r) {
            float v = fmaxf(fmaxf(sc[0][r], sc[1][r]), fmaxf(sc[2][r], sc[3][r]));
            v = red_max16(v);
            const float mn = fmaxf(mrun[r], v);
            alpha[r] = (mn == -__builtin_inff()) ? 1.f : __expf(mrun[r] - mn);
            mnew[r]  = mn;
        }

        float rs[8];
#pragma unroll
        for (int r = 0; r < 8; ++r) rs[r] = 0.f;
#pragma unroll
        for (int j = 0; j < 4; ++j) {
#pragma unroll
            for (int r = 0; r < 8; ++r) {
                const float sv = sc[j][r];
                const float e = (sv == -__builtin_inff()) ? 0.f
                                                          : __expf(sv - mnew[r]);
                rs[r] += e;
                pst[(r + kh * 8) * LSTRIDE + j * 16 + m] = (__bf16)e;
            }
        }
#pragma unroll
        for (int r = 0; r < 8; ++r) {
            lrun[r] = lrun[r] * alpha[r] + red_sum16(rs[r]);
            mrun[r] = mnew[r];
        }
#pragma unroll
        for (int t = 0; t < 4; ++t)
#pragma unroll
            for (int r = 0; r < 8; ++r)
                o[t][r] *= alpha[r];

        // ---- O += P @ V  (V^T in LDS -> contiguous B-fragments) ------------
        v16bf pa0, pa1;
        {
            const __bf16* pp = pst + m * LSTRIDE;
#pragma unroll
            for (int i = 0; i < 8; ++i) {
                pa0[i]     = pp[      kh * 8 + i];
                pa0[8 + i] = pp[16 +  kh * 8 + i];
                pa1[i]     = pp[32 +  kh * 8 + i];
                pa1[8 + i] = pp[48 +  kh * 8 + i];
            }
        }
#pragma unroll
        for (int t = 0; t < 4; ++t) {
            v16bf vb0, vb1;
            const __bf16* vp = Vl + (t * 16 + m) * LSTRIDE;   // row = dim
#pragma unroll
            for (int i = 0; i < 8; ++i) {
                vb0[i]     = vp[      kh * 8 + i];
                vb0[8 + i] = vp[16 +  kh * 8 + i];
                vb1[i]     = vp[32 +  kh * 8 + i];
                vb1[8 + i] = vp[48 +  kh * 8 + i];
            }
            o[t] = __builtin_amdgcn_wmma_f32_16x16x32_bf16(false, pa0, false, vb0,
                                                           (short)0, o[t], false, false);
            o[t] = __builtin_amdgcn_wmma_f32_16x16x32_bf16(false, pa1, false, vb1,
                                                           (short)0, o[t], false, false);
        }

        __syncthreads();   // all waves done with buf `cur` before TDM reuses it
    }

    // ---- epilogue: normalize and store fp32 --------------------------------
    float* Ob = O + (size_t)bh * SEQ * DH;
#pragma unroll
    for (int r = 0; r < 8; ++r) {
        const float inv = 1.f / lrun[r];
        const int qr = qbase + r + kh * 8;
#pragma unroll
        for (int t = 0; t < 4; ++t)
            Ob[(size_t)qr * DH + t * 16 + m] = o[t][r] * inv;
    }
}

// ---------------------------------------------------------------------------
// Fallback (no workspace): VMEM-staged variant, fp32 -> bf16 inline
// ---------------------------------------------------------------------------
__global__ __launch_bounds__(THREADS)
void fa_wmma_kernel(const float* __restrict__ Q,
                    const float* __restrict__ K,
                    const float* __restrict__ V,
                    const unsigned char* __restrict__ M,
                    float* __restrict__ O)
{
    __shared__ __bf16 Kl[KTILE * LSTRIDE];
    __shared__ __bf16 Vl[KTILE * LSTRIDE];
    __shared__ __bf16 Pl[NWAVE][16 * LSTRIDE];

    const int tid  = threadIdx.x;
    const int wave = tid >> 5;
    const int lane = tid & 31;
    const int m    = lane & 15;
    const int kh   = lane >> 4;

    const int qtile = blockIdx.x;
    const int bh    = blockIdx.y;
    const int qbase = qtile * QTILE + wave * 16;

    const float* Qb = Q + (size_t)bh * SEQ * DH;
    const float* Kb = K + (size_t)bh * SEQ * DH;
    const float* Vb = V + (size_t)bh * SEQ * DH;
    const unsigned char* Mb = M + (size_t)bh * SEQ * SEQ;

    v16bf qa0, qa1;
    {
        const float* qr = Qb + (size_t)(qbase + m) * DH;
#pragma unroll
        for (int i = 0; i < 8; ++i) {
            qa0[i]     = (__bf16)qr[      kh * 8 + i];
            qa0[8 + i] = (__bf16)qr[16 +  kh * 8 + i];
            qa1[i]     = (__bf16)qr[32 +  kh * 8 + i];
            qa1[8 + i] = (__bf16)qr[48 +  kh * 8 + i];
        }
    }

    float mrun[8], lrun[8];
    v8f   o[4];
    const v8f vzero = (v8f){0.f, 0.f, 0.f, 0.f, 0.f, 0.f, 0.f, 0.f};
#pragma unroll
    for (int r = 0; r < 8; ++r) { mrun[r] = -__builtin_inff(); lrun[r] = 0.f; }
#pragma unroll
    for (int t = 0; t < 4; ++t) o[t] = vzero;

    __bf16* pst = &Pl[wave][0];

    for (int kt = 0; kt < NTILES; ++kt) {
        const int kbase = kt * KTILE;

        __syncthreads();
        for (int i = tid; i < KTILE * DH; i += THREADS) {
            const int row = i >> 6, col = i & 63;
            Kl[row * LSTRIDE + col] = (__bf16)Kb[(size_t)(kbase + row) * DH + col];
            Vl[row * LSTRIDE + col] = (__bf16)Vb[(size_t)(kbase + row) * DH + col];
        }
        __syncthreads();

        if (kt + 1 < NTILES) {
            const char* nk = (const char*)(Kb + (size_t)(kbase + KTILE) * DH);
            __builtin_prefetch(nk + tid * 128, 0, 0);
        }

        v8f sc[4];
#pragma unroll
        for (int j = 0; j < 4; ++j) {
            v16bf kb0, kb1;
            const __bf16* kp = Kl + (j * 16 + m) * LSTRIDE;
#pragma unroll
            for (int i = 0; i < 8; ++i) {
                kb0[i]     = kp[      kh * 8 + i];
                kb0[8 + i] = kp[16 +  kh * 8 + i];
                kb1[i]     = kp[32 +  kh * 8 + i];
                kb1[8 + i] = kp[48 +  kh * 8 + i];
            }
            v8f acc = vzero;
            acc = __builtin_amdgcn_wmma_f32_16x16x32_bf16(false, qa0, false, kb0,
                                                          (short)0, acc, false, false);
            acc = __builtin_amdgcn_wmma_f32_16x16x32_bf16(false, qa1, false, kb1,
                                                          (short)0, acc, false, false);
            sc[j] = acc;
        }

#pragma unroll
        for (int j = 0; j < 4; ++j) {
            const int key = kbase + j * 16 + m;
            const unsigned char* mrow = Mb + key;
#pragma unroll
            for (int r = 0; r < 8; ++r) {
                const int qr = qbase + r + kh * 8;
                const unsigned char mk = mrow[(size_t)qr * SEQ];
                const float s = sc[j][r] * ATTN_SCALE;
                sc[j][r] = mk ? -__builtin_inff() : s;
            }
        }

        float mnew[8], alpha[8];
#pragma unroll
        for (int r = 0; r < 8; ++r) {
            float v = fmaxf(fmaxf(sc[0][r], sc[1][r]), fmaxf(sc[2][r], sc[3][r]));
            v = red_max16(v);
            const float mn = fmaxf(mrun[r], v);
            alpha[r] = (mn == -__builtin_inff()) ? 1.f : __expf(mrun[r] - mn);
            mnew[r]  = mn;
        }

        float rs[8];
#pragma unroll
        for (int r = 0; r < 8; ++r) rs[r] = 0.f;
#pragma unroll
        for (int j = 0; j < 4; ++j) {
#pragma unroll
            for (int r = 0; r < 8; ++r) {
                const float sv = sc[j][r];
                const float e = (sv == -__builtin_inff()) ? 0.f
                                                          : __expf(sv - mnew[r]);
                rs[r] += e;
                pst[(r + kh * 8) * LSTRIDE + j * 16 + m] = (__bf16)e;
            }
        }
#pragma unroll
        for (int r = 0; r < 8; ++r) {
            lrun[r] = lrun[r] * alpha[r] + red_sum16(rs[r]);
            mrun[r] = mnew[r];
        }
#pragma unroll
        for (int t = 0; t < 4; ++t)
#pragma unroll
            for (int r = 0; r < 8; ++r)
                o[t][r] *= alpha[r];

        v16bf pa0, pa1;
        {
            const __bf16* pp = pst + m * LSTRIDE;
#pragma unroll
            for (int i = 0; i < 8; ++i) {
                pa0[i]     = pp[      kh * 8 + i];
                pa0[8 + i] = pp[16 +  kh * 8 + i];
                pa1[i]     = pp[32 +  kh * 8 + i];
                pa1[8 + i] = pp[48 +  kh * 8 + i];
            }
        }
#pragma unroll
        for (int t = 0; t < 4; ++t) {
            v16bf vb0, vb1;
            const __bf16* vp = Vl + t * 16 + m;
#pragma unroll
            for (int i = 0; i < 8; ++i) {
                vb0[i]     = vp[(      kh * 8 + i) * LSTRIDE];
                vb0[8 + i] = vp[(16 +  kh * 8 + i) * LSTRIDE];
                vb1[i]     = vp[(32 +  kh * 8 + i) * LSTRIDE];
                vb1[8 + i] = vp[(48 +  kh * 8 + i) * LSTRIDE];
            }
            o[t] = __builtin_amdgcn_wmma_f32_16x16x32_bf16(false, pa0, false, vb0,
                                                           (short)0, o[t], false, false);
            o[t] = __builtin_amdgcn_wmma_f32_16x16x32_bf16(false, pa1, false, vb1,
                                                           (short)0, o[t], false, false);
        }
    }

    float* Ob = O + (size_t)bh * SEQ * DH;
#pragma unroll
    for (int r = 0; r < 8; ++r) {
        const float inv = 1.f / lrun[r];
        const int qr = qbase + r + kh * 8;
#pragma unroll
        for (int t = 0; t < 4; ++t)
            Ob[(size_t)qr * DH + t * 16 + m] = o[t][r] * inv;
    }
}

extern "C" void kernel_launch(void* const* d_in, const int* in_sizes, int n_in,
                              void* d_out, int out_size, void* d_ws, size_t ws_size,
                              hipStream_t stream) {
    const float*         q    = (const float*)d_in[0];
    const float*         k    = (const float*)d_in[1];
    const float*         v    = (const float*)d_in[2];
    const unsigned char* mask = (const unsigned char*)d_in[3];
    float*               out  = (float*)d_out;

    const size_t nkv  = (size_t)BH_N * SEQ * DH;          // 4,194,304 elements
    const size_t need = nkv * sizeof(__bf16) * 2;         // 16 MB workspace

    dim3 grid(SEQ / QTILE, BH_N);   // (32, 32) blocks

    if (ws_size >= need) {
        __bf16* kb  = (__bf16*)d_ws;
        __bf16* vbt = kb + nkv;
        const int cvt_blocks = (int)((nkv + 255) / 256);
        cvt_kv_kernel<<<cvt_blocks, 256, 0, stream>>>(k, v, kb, vbt);
        fa_wmma_tdm_kernel<<<grid, THREADS, 0, stream>>>(q, kb, vbt, mask, out);
    } else {
        fa_wmma_kernel<<<grid, THREADS, 0, stream>>>(q, k, v, mask, out);
    }
}
